// RWKV_TimeMix_63144609185888
// MI455X (gfx1250) — matrix operand order repaired and checked by reading, not compiled, not run on hardware
//
#include <hip/hip_runtime.h>
#include <hip/hip_bf16.h>

#define BS   8
#define TLEN 4096
#define CD   1024
#define NT   (BS * TLEN)      // 32768 rows
#define LDP  40               // padded LDS row stride (elements) -> 80B, 16B aligned
#define KSTEPS (CD / 32)      // 32 K-stages

typedef __attribute__((ext_vector_type(16))) __bf16         bf16x16;
typedef __attribute__((ext_vector_type(16))) unsigned short u16x16;
typedef __attribute__((ext_vector_type(8)))  unsigned short u16x8;
typedef __attribute__((ext_vector_type(4)))  unsigned short u16x4;
typedef __attribute__((ext_vector_type(8)))  float          f32x8;
typedef __attribute__((ext_vector_type(4)))  float          f32x4;

__device__ __forceinline__ unsigned short f2bf(float f) {
  unsigned int x = __builtin_bit_cast(unsigned int, f);
  x += 0x7fffu + ((x >> 16) & 1u);          // round-to-nearest-even
  return (unsigned short)(x >> 16);
}

// Async global -> LDS 16-byte copy (CDNA5, ASYNCcnt-tracked, GV addressing).
__device__ __forceinline__ void async_copy16(const unsigned short* gsrc,
                                             unsigned lds_off) {
  asm volatile("global_load_async_to_lds_b128 %0, %1, off"
               :: "v"(lds_off), "v"((unsigned long long)(uintptr_t)gsrc)
               : "memory");
}
__device__ __forceinline__ void async_wait0() {
  asm volatile("s_wait_asynccnt 0x0" ::: "memory");
}
__device__ __forceinline__ unsigned lds_off_of(const void* p) {
  return (unsigned)(uintptr_t)p;            // low 32 bits of shared address
}

// ---------------------------------------------------------------------------
// Pass 1: time-shift + mu-mix + fp32->bf16 pack of XR/XK/XV  (memory bound)
// ---------------------------------------------------------------------------
__global__ void __launch_bounds__(256) pack_mix(
    const float* __restrict__ x, const float* __restrict__ x_prev0,
    const float* __restrict__ mu_r, const float* __restrict__ mu_k,
    const float* __restrict__ mu_v,
    unsigned short* __restrict__ XR, unsigned short* __restrict__ XK,
    unsigned short* __restrict__ XV) {
  const int idx = blockIdx.x * blockDim.x + threadIdx.x;   // one 4-elem group
  const int n   = idx >> 8;                 // / (CD/4)
  const int c4  = (idx & 255) << 2;         // column*4
  const int b   = n / TLEN;
  const int t   = n & (TLEN - 1);

  f32x4 xc = *(const f32x4*)(x + (size_t)n * CD + c4);
  f32x4 xs = (t == 0) ? *(const f32x4*)(x_prev0 + (size_t)b * CD + c4)
                      : *(const f32x4*)(x + (size_t)(n - 1) * CD + c4);
  f32x4 mr = *(const f32x4*)(mu_r + c4);
  f32x4 mk = *(const f32x4*)(mu_k + c4);
  f32x4 mv = *(const f32x4*)(mu_v + c4);

  u16x4 or_, ok_, ov_;
#pragma unroll
  for (int i = 0; i < 4; ++i) {
    or_[i] = f2bf(mr[i] * xc[i] + (1.0f - mr[i]) * xs[i]);
    ok_[i] = f2bf(mk[i] * xc[i] + (1.0f - mk[i]) * xs[i]);
    ov_[i] = f2bf(mv[i] * xc[i] + (1.0f - mv[i]) * xs[i]);
  }
  *(u16x4*)(XR + (size_t)n * CD + c4) = or_;
  *(u16x4*)(XK + (size_t)n * CD + c4) = ok_;
  *(u16x4*)(XV + (size_t)n * CD + c4) = ov_;
}

// fp32 -> bf16 weight conversion
__global__ void __launch_bounds__(256) cvt_bf16(
    const float* __restrict__ src, unsigned short* __restrict__ dst) {
  const int i = (blockIdx.x * blockDim.x + threadIdx.x) << 2;
  f32x4 v = *(const f32x4*)(src + i);
  u16x4 o;
#pragma unroll
  for (int j = 0; j < 4; ++j) o[j] = f2bf(v[j]);
  *(u16x4*)(dst + i) = o;
}

// ---------------------------------------------------------------------------
// Fragment helpers (layouts per CDNA5 ISA §7.12.2, wave32)
// ---------------------------------------------------------------------------
__device__ __forceinline__ bf16x16 load_a_frag(const unsigned short* s, int mrow, int lane) {
  const int kb = (lane < 16) ? 0 : 8;
  u16x8 lo = *(const u16x8*)(s + mrow * LDP + kb);
  u16x8 hi = *(const u16x8*)(s + mrow * LDP + kb + 16);
  u16x16 u = __builtin_shufflevector(lo, hi, 0,1,2,3,4,5,6,7,8,9,10,11,12,13,14,15);
  return __builtin_bit_cast(bf16x16, u);
}

__device__ __forceinline__ bf16x16 load_b_frag(const unsigned short* s, int col, int lane) {
  const int kb = (lane < 16) ? 0 : 8;
  u16x8 lo = *(const u16x8*)(s + col * LDP + kb);
  u16x8 hi = *(const u16x8*)(s + col * LDP + kb + 16);
  u16x16 u = __builtin_shufflevector(lo, hi, 0,1,2,3,4,5,6,7,8,9,10,11,12,13,14,15);
  return __builtin_bit_cast(bf16x16, u);
}

__device__ __forceinline__ f32x8 wmma_bf16(bf16x16 a, bf16x16 b, f32x8 c) {
  return __builtin_amdgcn_wmma_f32_16x16x32_bf16(false, a, false, b,
                                                 (short)0, c, false, false);
}

// ---------------------------------------------------------------------------
// Pass 2: fused triple GEMM (r,k,v) + WKV epilogue -> RG = bf16(r * wkv)
// Block: 256 thr (8 waves, 4M x 2N). Tile 128(M) x 64(N). Wave: 32x32.
// Double-buffered LDS; async fills overlapped with WMMA compute.
// ---------------------------------------------------------------------------
__global__ void __launch_bounds__(256) gemm_rkv_wkv(
    const unsigned short* __restrict__ XR, const unsigned short* __restrict__ XK,
    const unsigned short* __restrict__ XV,
    const unsigned short* __restrict__ Wr, const unsigned short* __restrict__ Wk,
    const unsigned short* __restrict__ Wv,
    const float* __restrict__ w, const float* __restrict__ u,
    const float* __restrict__ a0, const float* __restrict__ b0,
    unsigned short* __restrict__ RG) {
  __shared__ unsigned short As[2][3][128 * LDP];   // 2 x 3 x 10240 B
  __shared__ unsigned short Bs[2][3][64 * LDP];    // 2 x 3 x  5120 B

  const int tid  = threadIdx.x;
  const int lane = tid & 31;
  const int wave = tid >> 5;
  const int wm   = wave & 3;            // M block (32 rows)
  const int wn   = wave >> 2;           // N block (32 cols)
  const int nT   = blockIdx.x * 128;
  const int dT   = blockIdx.y * 64;

  const unsigned short* Ag[3] = {XR, XK, XV};
  const unsigned short* Bg[3] = {Wr, Wk, Wv};

  f32x8 acc[3][2][2] = {};

  const int aRow = tid >> 2, aC8 = (tid & 3) * 8;   // fill coordinates

  // Prologue: fill stage 0 into buffer 0
#pragma unroll
  for (int q = 0; q < 3; ++q) {
#pragma unroll
    for (int h = 0; h < 2; ++h) {
      const int rc = tid + h * 256, row = rc >> 2, c8 = (rc & 3) * 8;
      async_copy16(Ag[q] + (size_t)(nT + row) * CD + c8,
                   lds_off_of(&As[0][q][row * LDP + c8]));
    }
    async_copy16(Bg[q] + (size_t)(dT + aRow) * CD + aC8,
                 lds_off_of(&Bs[0][q][aRow * LDP + aC8]));
  }

#pragma unroll 2
  for (int it = 0; it < KSTEPS; ++it) {
    const int buf = it & 1;
    async_wait0();          // this wave's fills for `buf` are done
    __syncthreads();        // everyone's fills done; prev reads of buf^1 done

    if (it + 1 < KSTEPS) {  // issue next stage into other buffer (overlapped)
      const int kn = (it + 1) * 32;
#pragma unroll
      for (int q = 0; q < 3; ++q) {
#pragma unroll
        for (int h = 0; h < 2; ++h) {
          const int rc = tid + h * 256, row = rc >> 2, c8 = (rc & 3) * 8;
          async_copy16(Ag[q] + (size_t)(nT + row) * CD + kn + c8,
                       lds_off_of(&As[buf ^ 1][q][row * LDP + c8]));
        }
        async_copy16(Bg[q] + (size_t)(dT + aRow) * CD + kn + aC8,
                     lds_off_of(&Bs[buf ^ 1][q][aRow * LDP + aC8]));
      }
    }

#pragma unroll
    for (int q = 0; q < 3; ++q) {
      bf16x16 a0f = load_a_frag(As[buf][q], wm * 32 +      (lane & 15), lane);
      bf16x16 a1f = load_a_frag(As[buf][q], wm * 32 + 16 + (lane & 15), lane);
      bf16x16 b0f = load_b_frag(Bs[buf][q], wn * 32 +      (lane & 15), lane);
      bf16x16 b1f = load_b_frag(Bs[buf][q], wn * 32 + 16 + (lane & 15), lane);
      acc[q][0][0] = wmma_bf16(a0f, b0f, acc[q][0][0]);
      acc[q][0][1] = wmma_bf16(a0f, b1f, acc[q][0][1]);
      acc[q][1][0] = wmma_bf16(a1f, b0f, acc[q][1][0]);
      acc[q][1][1] = wmma_bf16(a1f, b1f, acc[q][1][1]);
    }
  }

  // WKV epilogue (elementwise in (n, d); block stays inside one batch b)
  const int bb = nT / TLEN;
#pragma unroll
  for (int m = 0; m < 2; ++m) {
    const int rowBase = nT + wm * 32 + m * 16 + ((lane < 16) ? 0 : 8);
#pragma unroll
    for (int nn = 0; nn < 2; ++nn) {
      const int   col = dT + wn * 32 + nn * 16 + (lane & 15);
      const float uc  = u[col];
      const float wc  = w[col];
      const float a0c = a0[bb * CD + col];
      const float b0c = b0[bb * CD + col];
#pragma unroll
      for (int i = 0; i < 8; ++i) {
        const int n = rowBase + i;
        const int t = n & (TLEN - 1);
        float r   = 1.0f / (1.0f + __expf(-acc[0][m][nn][i]));
        float ek  = __expf(uc + acc[1][m][nn][i]);
        float dp  = __expf(-wc * (float)(t + 1));
        float wkv = (a0c * dp + ek * acc[2][m][nn][i]) / (b0c * dp + ek + 1e-8f);
        RG[(size_t)n * CD + col] = f2bf(r * wkv);
      }
    }
  }
}

// ---------------------------------------------------------------------------
// Pass 3: out = RG @ W_out^T  (fp32 output into d_out)
// Block: 256 thr (8 waves, 4M x 2N). Tile 128(M) x 128(N). Wave: 32x64.
// ---------------------------------------------------------------------------
__global__ void __launch_bounds__(256) gemm_out(
    const unsigned short* __restrict__ RG, const unsigned short* __restrict__ Wo,
    float* __restrict__ out) {
  __shared__ unsigned short As[2][128 * LDP];
  __shared__ unsigned short Bs[2][128 * LDP];

  const int tid  = threadIdx.x;
  const int lane = tid & 31;
  const int wave = tid >> 5;
  const int wm   = wave & 3;            // M block (32 rows)
  const int wn   = wave >> 2;           // N block (64 cols)
  const int nT   = blockIdx.x * 128;
  const int dT   = blockIdx.y * 128;

  f32x8 acc[2][4] = {};

  // Prologue fill (stage 0, buffer 0)
#pragma unroll
  for (int h = 0; h < 2; ++h) {
    const int rc = tid + h * 256, row = rc >> 2, c8 = (rc & 3) * 8;
    async_copy16(RG + (size_t)(nT + row) * CD + c8,
                 lds_off_of(&As[0][row * LDP + c8]));
    async_copy16(Wo + (size_t)(dT + row) * CD + c8,
                 lds_off_of(&Bs[0][row * LDP + c8]));
  }

#pragma unroll 2
  for (int it = 0; it < KSTEPS; ++it) {
    const int buf = it & 1;
    async_wait0();
    __syncthreads();

    if (it + 1 < KSTEPS) {
      const int kn = (it + 1) * 32;
#pragma unroll
      for (int h = 0; h < 2; ++h) {
        const int rc = tid + h * 256, row = rc >> 2, c8 = (rc & 3) * 8;
        async_copy16(RG + (size_t)(nT + row) * CD + kn + c8,
                     lds_off_of(&As[buf ^ 1][row * LDP + c8]));
        async_copy16(Wo + (size_t)(dT + row) * CD + kn + c8,
                     lds_off_of(&Bs[buf ^ 1][row * LDP + c8]));
      }
    }

    bf16x16 a0f = load_a_frag(As[buf], wm * 32 +      (lane & 15), lane);
    bf16x16 a1f = load_a_frag(As[buf], wm * 32 + 16 + (lane & 15), lane);
#pragma unroll
    for (int nn = 0; nn < 4; ++nn) {
      bf16x16 bf = load_b_frag(Bs[buf], wn * 64 + nn * 16 + (lane & 15), lane);
      acc[0][nn] = wmma_bf16(a0f, bf, acc[0][nn]);
      acc[1][nn] = wmma_bf16(a1f, bf, acc[1][nn]);
    }
  }

#pragma unroll
  for (int m = 0; m < 2; ++m) {
    const int rowBase = nT + wm * 32 + m * 16 + ((lane < 16) ? 0 : 8);
#pragma unroll
    for (int nn = 0; nn < 4; ++nn) {
      const int col = dT + wn * 64 + nn * 16 + (lane & 15);
#pragma unroll
      for (int i = 0; i < 8; ++i)
        out[(size_t)(rowBase + i) * CD + col] = acc[m][nn][i];
    }
  }
}

// ---------------------------------------------------------------------------
// Tail: a_T, b_T, x_last
// ---------------------------------------------------------------------------
__global__ void __launch_bounds__(256) tail(
    const float* __restrict__ x, const float* __restrict__ a0,
    const float* __restrict__ b0, const float* __restrict__ w,
    float* __restrict__ out) {
  const int i = blockIdx.x * blockDim.x + threadIdx.x;
  if (i >= BS * CD) return;
  const int b = i / CD, c = i & (CD - 1);
  const float dp = __expf(-w[c] * (float)TLEN);
  float* aT = out + (size_t)NT * CD;
  float* bT = aT + BS * CD;
  float* xl = bT + BS * CD;
  aT[i] = a0[i] * dp;
  bT[i] = b0[i] * dp;
  xl[i] = x[((size_t)b * TLEN + (TLEN - 1)) * CD + c];
}

// ---------------------------------------------------------------------------
extern "C" void kernel_launch(void* const* d_in, const int* in_sizes, int n_in,
                              void* d_out, int out_size, void* d_ws, size_t ws_size,
                              hipStream_t stream) {
  const float* x       = (const float*)d_in[0];
  const float* a0      = (const float*)d_in[1];
  const float* b0      = (const float*)d_in[2];
  const float* x_prev0 = (const float*)d_in[3];
  const float* w       = (const float*)d_in[4];
  const float* u       = (const float*)d_in[5];
  const float* mu_r    = (const float*)d_in[6];
  const float* mu_k    = (const float*)d_in[7];
  const float* mu_v    = (const float*)d_in[8];
  const float* W_r     = (const float*)d_in[9];
  const float* W_k     = (const float*)d_in[10];
  const float* W_v     = (const float*)d_in[11];
  const float* W_out   = (const float*)d_in[12];
  float* out = (float*)d_out;

  const size_t matE = (size_t)NT * CD;       // 33.55M elements
  unsigned short* ws  = (unsigned short*)d_ws;
  unsigned short* XR  = ws;
  unsigned short* XK  = XR + matE;
  unsigned short* XV  = XK + matE;
  unsigned short* RG  = XV + matE;
  unsigned short* WrB = RG + matE;
  unsigned short* WkB = WrB + (size_t)CD * CD;
  unsigned short* WvB = WkB + (size_t)CD * CD;
  unsigned short* WoB = WvB + (size_t)CD * CD;

  // Pass 0: weight conversion (4 x 1M elements, 4 per thread)
  const int cvtBlocks = (CD * CD / 4) / 256;
  cvt_bf16<<<cvtBlocks, 256, 0, stream>>>(W_r,   WrB);
  cvt_bf16<<<cvtBlocks, 256, 0, stream>>>(W_k,   WkB);
  cvt_bf16<<<cvtBlocks, 256, 0, stream>>>(W_v,   WvB);
  cvt_bf16<<<cvtBlocks, 256, 0, stream>>>(W_out, WoB);

  // Pass 1: mix + pack
  pack_mix<<<(NT * (CD / 4)) / 256, 256, 0, stream>>>(
      x, x_prev0, mu_r, mu_k, mu_v, XR, XK, XV);

  // Pass 2: fused r/k/v GEMMs + wkv epilogue
  dim3 grid2(NT / 128, CD / 64);
  gemm_rkv_wkv<<<grid2, 256, 0, stream>>>(XR, XK, XV, WrB, WkB, WvB,
                                          w, u, a0, b0, RG);

  // Pass 3: output projection
  dim3 grid3(NT / 128, CD / 128);
  gemm_out<<<grid3, 256, 0, stream>>>(RG, WoB, out);

  // Tail outputs
  tail<<<(BS * CD + 255) / 256, 256, 0, stream>>>(x, a0, b0, w, out);
}
